// LSTM_35562329211204
// MI455X (gfx1250) — matrix operand order repaired and checked by reading, not compiled, not run on hardware
//
#include <hip/hip_runtime.h>
#include <hip/hip_bf16.h>
#include <stdint.h>

// ---------------------------------------------------------------------------
// LSTM forward for MI455X (gfx1250, wave32).
//   K0a: Why [512,32000] f32 -> WhyT [32000,512] bf16 (LDS-tiled transpose)
//   K0b: init h ping-pong / c / grid-barrier counter
//   K1 : embedding gather + 4 input projections, V_WMMA_F32_16X16X4_F32
//   K2 : persistent 256-step recurrence, 32 WGs x 4 waves (1 wave per gate),
//        h staged to LDS via async-load-to-LDS (ASYNCcnt), fp32 WMMA,
//        LDS gate exchange, 1 grid barrier per step (ping-pong h)
//   K3 : output projection bf16 WMMA, 32x64 per wave, B frags shared across
//        the block via L0, non-temporal stores for the 524 MB output stream
// ---------------------------------------------------------------------------

typedef __attribute__((ext_vector_type(2)))  float        v2f;
typedef __attribute__((ext_vector_type(4)))  float        v4f;
typedef __attribute__((ext_vector_type(8)))  float        v8f;
typedef __attribute__((ext_vector_type(16))) __bf16       v16bf;
typedef __attribute__((ext_vector_type(4)))  unsigned int v4u;
// exact pointee type expected by the async-load-to-LDS builtin (from hipcc diag)
typedef int v4i_vs __attribute__((__vector_size__(4 * sizeof(int))));
typedef __attribute__((address_space(1))) v4i_vs* g_v4i_p;   // global (AS1)
typedef __attribute__((address_space(3))) v4i_vs* l_v4i_p;   // LDS    (AS3)

#define VOCAB 32000
#define EDIM  128
#define HDIM  512
#define ODIM  32000
#define BB    16
#define SS    256
#define BSROWS (BB*SS)          /* 4096 */
#define NWG_REC 32              /* workgroups in the recurrence kernel */

#define OUT_H_OFF (BB*SS*ODIM)          /* 131072000 */
#define OUT_C_OFF (OUT_H_OFF + BB*HDIM) /* 131080192 */

// workspace byte offsets
#define WS_XG    0u                       /* 4*4096*512*4  = 33554432 */
#define WS_HBUF  33554432u                /* 2*16*512*4    = 65536    */
#define WS_CBUF  33619968u                /* 16*512*4      = 32768    */
#define WS_HSBF  33652736u                /* 4096*512*2    = 4194304  */
#define WS_WHYT  37847040u                /* 32000*512*2   = 32768000 */
#define WS_BAR   70615040u                /* 4 bytes                   */

#if __has_builtin(__builtin_amdgcn_global_load_async_to_lds_b128)
#define HAVE_ASYNC_LDS 1
#else
#define HAVE_ASYNC_LDS 0
#endif

__device__ __forceinline__ unsigned short f2bf(float f) {
    union { float f; unsigned u; } x; x.f = f;
    unsigned u = x.u;
    unsigned r = (u + 0x7FFFu + ((u >> 16) & 1u)) >> 16;   // round-nearest-even
    return (unsigned short)r;
}

// ---- K0a: Why -> WhyT (bf16) via LDS-tiled transpose (coalesced both sides).
//      grid: 16 k-tiles x 500 n-tiles = 8000 blocks, 256 threads.
__global__ void k_convert_whyT(const float* __restrict__ why,
                               unsigned short* __restrict__ whyT) {
    __shared__ float t[32][65];                 // +1 pad: conflict-free columns
    const unsigned kb = blockIdx.x / 500u;      // 32-row k tile
    const unsigned nb = blockIdx.x % 500u;      // 64-col n tile
#pragma unroll
    for (unsigned i = threadIdx.x; i < 32u * 64u; i += 256u) {
        const unsigned kk = i >> 6, nn = i & 63u;
        t[kk][nn] = why[(size_t)(kb * 32u + kk) * ODIM + nb * 64u + nn];
    }
    __syncthreads();
    const unsigned n_local = threadIdx.x >> 2;          // 0..63
    const unsigned kc      = (threadIdx.x & 3u) * 8u;   // 0,8,16,24
    unsigned short tmp[8];
#pragma unroll
    for (int j = 0; j < 8; ++j) tmp[j] = f2bf(t[kc + j][n_local]);
    v4u pack;
    pack.x = (unsigned)tmp[0] | ((unsigned)tmp[1] << 16);
    pack.y = (unsigned)tmp[2] | ((unsigned)tmp[3] << 16);
    pack.z = (unsigned)tmp[4] | ((unsigned)tmp[5] << 16);
    pack.w = (unsigned)tmp[6] | ((unsigned)tmp[7] << 16);
    *(v4u*)(whyT + (size_t)(nb * 64u + n_local) * HDIM + kb * 32u + kc) = pack;
}

// ---- K0b: per-launch state init (ws is NOT re-poisoned between replays).
__global__ void k_init(const float* __restrict__ h0, const float* __restrict__ c0,
                       float* __restrict__ hbuf, float* __restrict__ cbuf,
                       unsigned* __restrict__ bar) {
    unsigned i = blockIdx.x * blockDim.x + threadIdx.x;     // 8192 threads
    hbuf[i] = h0[i];            // ping half 0
    cbuf[i] = c0[i];
    if (i == 0) *bar = 0u;
}

// ---- K1: xg[gate] = gather(emb, x) @ Wx[gate] + bx[gate], fp32 WMMA.
__global__ void k_input_proj(const int* __restrict__ x, const float* __restrict__ emb,
                             const float* __restrict__ Wxi, const float* __restrict__ Wxf,
                             const float* __restrict__ Wxc, const float* __restrict__ Wxo,
                             const float* __restrict__ bxi, const float* __restrict__ bxf,
                             const float* __restrict__ bxc, const float* __restrict__ bxo,
                             float* __restrict__ xg) {
    const unsigned wid  = blockIdx.x * 8u + (threadIdx.x >> 5);   // 32768 wave-tiles
    const unsigned lane = threadIdx.x & 31u;
    const unsigned mtile = wid >> 7;          // 0..255
    const unsigned rem   = wid & 127u;
    const unsigned gate  = rem >> 5;          // 0..3
    const unsigned ntile = rem & 31u;         // 0..31

    const float* W    = (gate == 0) ? Wxi : (gate == 1) ? Wxf : (gate == 2) ? Wxc : Wxo;
    const float* bias = (gate == 0) ? bxi : (gate == 1) ? bxf : (gate == 2) ? bxc : bxo;

    const unsigned n     = lane & 15u;
    const unsigned khalf = lane >> 4;
    const unsigned row   = mtile * 16u + n;            // A-row for this lane (M = lane%16)
    const int tok        = x[row];
    const float* arow    = emb + (size_t)tok * EDIM;
    const unsigned col   = ntile * 16u + n;            // B-col / D-col (N = lane%16)

    v8f acc = {};
#pragma unroll 8
    for (int k0 = 0; k0 < EDIM; k0 += 4) {
        const int ka = k0 + 2 * (int)khalf;
        v2f a; a.x = arow[ka];            a.y = arow[ka + 1];
        v2f b; b.x = W[ka * HDIM + col];  b.y = W[(ka + 1) * HDIM + col];
        acc = __builtin_amdgcn_wmma_f32_16x16x4_f32(false, a, false, b,
                                                    (short)0, acc, false, false);
    }
    const float bv = bias[col];
    float* dst = xg + (size_t)gate * BSROWS * HDIM;
#pragma unroll
    for (int r = 0; r < 8; ++r) {
        const unsigned rr = mtile * 16u + (unsigned)r + 8u * khalf;   // D: M = r + 8*(L/16)
        dst[(size_t)rr * HDIM + col] = acc[r] + bv;
    }
}

// ---- K2: persistent recurrence. WG wg owns h-columns [wg*16, wg*16+16);
//          wave g (=gate) computes the [16 x 16] preact tile for its gate.
//          h is staged to LDS once per step (async-load-to-LDS when available)
//          because the A operand is identical for all 4 gate-waves.
__global__ void k_recurrence(const float* __restrict__ xg,
                             float* __restrict__ hbuf,    /* [2][16][512] ping-pong */
                             float* __restrict__ cbuf,    /* [16][512]              */
                             const float* __restrict__ Whi, const float* __restrict__ Whf,
                             const float* __restrict__ Whc, const float* __restrict__ Who,
                             unsigned short* __restrict__ hsbf, /* [S*B][512] bf16   */
                             float* __restrict__ dout,
                             unsigned* __restrict__ bar) {
    __shared__ float hsh[BB * HDIM];    // 32 KB staged h(t)
    __shared__ float gbuf[4][16][16];   // gate preacts exchanged inside the WG

    const unsigned wg   = blockIdx.x;          // 0..31
    const unsigned wave = threadIdx.x >> 5;    // 0..3  == gate
    const unsigned lane = threadIdx.x & 31u;
    const unsigned n     = lane & 15u;
    const unsigned khalf = lane >> 4;
    const unsigned col   = wg * 16u + n;

    const float* W = (wave == 0) ? Whi : (wave == 1) ? Whf : (wave == 2) ? Whc : Who;
    const float* xgg = xg + (size_t)wave * BSROWS * HDIM;

    for (int s = 0; s < SS; ++s) {
        // ---- stage h(t) (16x512 f32 = 2048 16B chunks) into LDS
        const float* hsrc = hbuf + (size_t)(s & 1) * BB * HDIM;
        for (unsigned chunk = threadIdx.x; chunk < (BB * HDIM / 4u); chunk += 128u) {
#if HAVE_ASYNC_LDS
            __builtin_amdgcn_global_load_async_to_lds_b128(
                (g_v4i_p)(uintptr_t)(hsrc + chunk * 4u),
                (l_v4i_p)(unsigned)(uintptr_t)(hsh + chunk * 4u),
                0, 0);
#else
            *(v4f*)(hsh + chunk * 4u) = *(const v4f*)(hsrc + chunk * 4u);
#endif
        }
#if HAVE_ASYNC_LDS
#if __has_builtin(__builtin_amdgcn_s_wait_asynccnt)
        __builtin_amdgcn_s_wait_asynccnt(0);
#else
        asm volatile("s_wait_asynccnt 0x0" ::: "memory");
#endif
#endif
        __syncthreads();

        // ---- gate GEMM: [16,512] (LDS) x [512,16] (L2-resident Wh slice)
        const unsigned arow = n * HDIM;        // A-row = batch index (M = lane%16)
        v8f acc = {};
#pragma unroll 4
        for (int k0 = 0; k0 < HDIM; k0 += 4) {
            const int ka = k0 + 2 * (int)khalf;
            v2f a; a.x = hsh[arow + ka];      a.y = hsh[arow + ka + 1];
            v2f b; b.x = W[ka * HDIM + col];  b.y = W[(ka + 1) * HDIM + col];
            acc = __builtin_amdgcn_wmma_f32_16x16x4_f32(false, a, false, b,
                                                        (short)0, acc, false, false);
        }
        // add precomputed input projection, publish to LDS
#pragma unroll
        for (int r = 0; r < 8; ++r) {
            const unsigned bb = (unsigned)r + 8u * khalf;               // batch
            gbuf[wave][bb][n] = acc[r] + xgg[((size_t)bb * SS + s) * HDIM + col];
        }
        __syncthreads();

        // ---- element-wise LSTM cell update: 256 elems over 128 threads
        for (unsigned e = threadIdx.x; e < 256u; e += 128u) {
            const unsigned bb = e >> 4, nn = e & 15u, cc = wg * 16u + nn;
            const float pi = gbuf[0][bb][nn];
            const float pf = gbuf[1][bb][nn];
            const float pc = gbuf[2][bb][nn];
            const float po = gbuf[3][bb][nn];
            const float ig = 1.0f / (1.0f + expf(-pi));
            const float fg = 1.0f / (1.0f + expf(-pf));
            const float og = 1.0f / (1.0f + expf(-po));
            const float gg = tanhf(pc);
            const float cold = cbuf[bb * HDIM + cc];
            const float cn = fg * cold + ig * gg;
            const float hn = og * tanhf(cn);
            cbuf[bb * HDIM + cc] = cn;
            hbuf[(size_t)((s + 1) & 1) * BB * HDIM + bb * HDIM + cc] = hn;
            hsbf[((size_t)s * BB + bb) * HDIM + cc] = f2bf(hn);
            if (s == SS - 1) {
                dout[OUT_H_OFF + bb * HDIM + cc] = hn;
                dout[OUT_C_OFF + bb * HDIM + cc] = cn;
            }
        }

        // ---- one grid-wide barrier per step (monotonic counter)
        __syncthreads();
        if (threadIdx.x == 0) {
            __threadfence();
            __hip_atomic_fetch_add(bar, 1u, __ATOMIC_RELEASE, __HIP_MEMORY_SCOPE_AGENT);
            const unsigned tgt = NWG_REC * (unsigned)(s + 1);
            while (__hip_atomic_load(bar, __ATOMIC_ACQUIRE, __HIP_MEMORY_SCOPE_AGENT) < tgt)
                __builtin_amdgcn_s_sleep(2);
        }
        __syncthreads();
    }
}

// ---- K3: logits = hs @ Why + bhy, bf16 WMMA, 32x64 per wave.
//      All 8 waves of a block use the same nblk -> identical B frags (L0 hits),
//      adjacent mblks -> streaming A. Non-temporal stores for the output.
__global__ void k_out_proj(const unsigned short* __restrict__ hsbf,
                           const unsigned short* __restrict__ whyT,
                           const float* __restrict__ bhy,
                           float* __restrict__ out) {
    const unsigned wslot = threadIdx.x >> 5;
    const unsigned lane  = threadIdx.x & 31u;
    const unsigned mblk  = (blockIdx.x / 500u) * 8u + wslot;  // 0..127 (32 rows each)
    const unsigned nblk  = blockIdx.x % 500u;                 // 0..499 (64 cols each)
    const unsigned n     = lane & 15u;
    const unsigned khalf = lane >> 4;

    const unsigned short* ap[2];
    const unsigned short* bp[4];
#pragma unroll
    for (int mi = 0; mi < 2; ++mi)
        ap[mi] = hsbf + (size_t)(mblk * 32u + mi * 16u + n) * HDIM;
#pragma unroll
    for (int ni = 0; ni < 4; ++ni)
        bp[ni] = whyT + (size_t)(nblk * 64u + ni * 16u + n) * HDIM;

    v8f acc[2][4] = {};
#pragma unroll 4
    for (int k0 = 0; k0 < HDIM; k0 += 32) {
        union { v4u q[2]; v16bf v; } ua[2], ub[4];
#pragma unroll
        for (int mi = 0; mi < 2; ++mi) {
            // A 16x32 bf16: elems j -> K=k0+8*khalf+j ; 8+j -> K=k0+16+8*khalf+j
            ua[mi].q[0] = *(const v4u*)(ap[mi] + k0 + 8 * khalf);
            ua[mi].q[1] = *(const v4u*)(ap[mi] + k0 + 16 + 8 * khalf);
        }
#pragma unroll
        for (int ni = 0; ni < 4; ++ni) {
            // B 32x16 bf16: elems j -> K=k0+8*khalf+j ; 8+j -> K=k0+16+8*khalf+j
            ub[ni].q[0] = *(const v4u*)(bp[ni] + k0 + 8 * khalf);
            ub[ni].q[1] = *(const v4u*)(bp[ni] + k0 + 16 + 8 * khalf);
        }
#pragma unroll
        for (int mi = 0; mi < 2; ++mi)
#pragma unroll
            for (int ni = 0; ni < 4; ++ni)
                acc[mi][ni] = __builtin_amdgcn_wmma_f32_16x16x32_bf16(
                    false, ua[mi].v, false, ub[ni].v, (short)0, acc[mi][ni],
                    false, false);
    }

#pragma unroll
    for (int ni = 0; ni < 4; ++ni) {
        const unsigned col = nblk * 64u + ni * 16u + n;
        const float bv = bhy[col];
#pragma unroll
        for (int mi = 0; mi < 2; ++mi) {
#pragma unroll
            for (int r = 0; r < 8; ++r) {
                const unsigned hsrow = mblk * 32u + mi * 16u + (unsigned)r + 8u * khalf;
                const unsigned b = hsrow & 15u, s = hsrow >> 4;   // hs row = s*16 + b
                __builtin_nontemporal_store(acc[mi][ni][r] + bv,
                                            &out[((size_t)b * SS + s) * ODIM + col]);
            }
        }
    }
}

extern "C" void kernel_launch(void* const* d_in, const int* in_sizes, int n_in,
                              void* d_out, int out_size, void* d_ws, size_t ws_size,
                              hipStream_t stream) {
    (void)in_sizes; (void)n_in; (void)out_size;
    const int*   x    = (const int*)  d_in[0];
    const float* h0   = (const float*)d_in[1];
    const float* c0   = (const float*)d_in[2];
    const float* emb  = (const float*)d_in[3];
    const float* Wxi  = (const float*)d_in[4];
    const float* bxi  = (const float*)d_in[5];
    const float* Whi  = (const float*)d_in[6];
    const float* Wxf  = (const float*)d_in[7];
    const float* bxf  = (const float*)d_in[8];
    const float* Whf  = (const float*)d_in[9];
    const float* Wxc  = (const float*)d_in[10];
    const float* bxc  = (const float*)d_in[11];
    const float* Whc  = (const float*)d_in[12];
    const float* Wxo  = (const float*)d_in[13];
    const float* bxo  = (const float*)d_in[14];
    const float* Who  = (const float*)d_in[15];
    const float* Why  = (const float*)d_in[16];
    const float* bhy  = (const float*)d_in[17];

    char* ws = (char*)d_ws;                    // needs ~70.7 MB
    float*          xg   = (float*)(ws + WS_XG);
    float*          hbuf = (float*)(ws + WS_HBUF);
    float*          cbuf = (float*)(ws + WS_CBUF);
    unsigned short* hsbf = (unsigned short*)(ws + WS_HSBF);
    unsigned short* whyT = (unsigned short*)(ws + WS_WHYT);
    unsigned*       bar  = (unsigned*)(ws + WS_BAR);
    (void)ws_size;

    float* out = (float*)d_out;

    k_convert_whyT<<<8000, 256, 0, stream>>>(Why, whyT);
    k_init        <<<32,   256, 0, stream>>>(h0, c0, hbuf, cbuf, bar);
    k_input_proj  <<<4096, 256, 0, stream>>>(x, emb, Wxi, Wxf, Wxc, Wxo,
                                             bxi, bxf, bxc, bxo, xg);
    k_recurrence  <<<NWG_REC, 128, 0, stream>>>(xg, hbuf, cbuf,
                                                Whi, Whf, Whc, Who,
                                                hsbf, out, bar);
    k_out_proj    <<<8000, 256, 0, stream>>>(hsbf, whyT, bhy, out);
}
